// TemporalEmbeddingLoss_20083267076319
// MI455X (gfx1250) — matrix operand order repaired and checked by reading, not compiled, not run on hardware
//
#include <hip/hip_runtime.h>
#include <hip/hip_bf16.h>

typedef __attribute__((ext_vector_type(16))) _Float16 v16h;
typedef __attribute__((ext_vector_type(8)))  float    v8f;
typedef __attribute__((ext_vector_type(4)))  float    v4f;
typedef __attribute__((ext_vector_type(4)))  int      v4i;

#define B_  4
#define F_  8
#define C_  16
#define H_  256
#define W_  256
#define HW_ (H_ * W_)            // 65536 pixels per slab
#define K_  64                   // NUM_IDS
#define SLABS (B_ * F_)          // 32
#define BLOCKS_PER_SLAB 16
#define PIX_PER_BLOCK (HW_ / BLOCKS_PER_SLAB)   // 4096
#define THREADS 256
#define WAVES   (THREADS / 32)
#define WS_SUM_ELEMS (SLABS * K_ * C_)          // 32768 floats
#define WS_CNT_ELEMS (SLABS * K_)               // 2048 ints
#define WS_TOTAL_WORDS (WS_SUM_ELEMS + WS_CNT_ELEMS)

// ---------------------------------------------------------------------------
// Kernel 0: zero the workspace (sums + counts) — must run every launch since
// the harness does not re-poison/zero between graph replays.
// ---------------------------------------------------------------------------
__global__ __launch_bounds__(256) void zero_ws_kernel(unsigned* __restrict__ ws, int n) {
    int i = blockIdx.x * blockDim.x + threadIdx.x;
    if (i < n) ws[i] = 0u;
}

// ---------------------------------------------------------------------------
// Kernel 1: per-slab segmented sums via WMMA one-hot GEMM + counts via LDS.
//   sums(64 ids x 16 ch) = OneHot(64 x Npix) * Emb(Npix x 16)
// Each wave iteration consumes 32 pixels with 4x v_wmma_f32_16x16x32_f16
// (one per 16-id tile), accumulating in f32. The four A tiles are pinned
// into four DISTINCT register tuples (asm "+v" + sched_barrier) before the
// WMMA burst: back-to-back WMMAs then share no A/B/D registers (no hazard
// NOPs between them) and the next iteration's one-hot rebuild lands >= 3
// WMMAs after the WMMA that read those registers, absorbing the 4-coexec
// WAR window the hardware requires for F16 WMMA.
// ---------------------------------------------------------------------------
__global__ __launch_bounds__(256) void accum_kernel(const float* __restrict__ emb,
                                                    const int*   __restrict__ ids,
                                                    float* __restrict__ g_sums,
                                                    int*   __restrict__ g_counts) {
    __shared__ float lds_sums[K_ * C_];
    __shared__ int   lds_counts[K_];

    const int tid = threadIdx.x;
    for (int i = tid; i < K_ * C_; i += THREADS) lds_sums[i] = 0.0f;
    if (tid < K_) lds_counts[tid] = 0;
    __syncthreads();

    const int slab = blockIdx.x / BLOCKS_PER_SLAB;
    const int blk  = blockIdx.x % BLOCKS_PER_SLAB;

    const float* eslab = emb + (size_t)slab * C_ * HW_;
    const int*   islab = ids + (size_t)slab * HW_;

    const int lane = tid & 31;          // wave32
    const int wave = tid >> 5;
    const int half = lane >> 4;         // 0: lanes 0-15, 1: lanes 16-31
    const int col  = lane & 15;         // channel / matrix row-in-tile

    v8f acc0 = (v8f){}, acc1 = (v8f){}, acc2 = (v8f){}, acc3 = (v8f){};

    const _Float16 one  = (_Float16)1.0f;
    const _Float16 zero = (_Float16)0.0f;

    const int t0 = 0  + col;            // per-tile one-hot targets
    const int t1 = 16 + col;
    const int t2 = 32 + col;
    const int t3 = 48 + col;

    const int base0  = blk * PIX_PER_BLOCK;
    const int niters = PIX_PER_BLOCK / (WAVES * 32);   // 16
    const int K0     = half * 16;

    for (int it = 0; it < niters; ++it) {
        const int pbase = base0 + it * (WAVES * 32) + wave * 32;

        // ---- B matrix (32x16 f16): lane holds channel `col`,
        //      element j <-> K = 16*half + j  (contiguous pixels)
        const float* ep = eslab + (size_t)col * HW_ + pbase + half * 16;
        v4f e0 = *(const v4f*)(ep + 0);
        v4f e1 = *(const v4f*)(ep + 4);
        v4f e2 = *(const v4f*)(ep + 8);
        v4f e3 = *(const v4f*)(ep + 12);
        float ev[16] = {e0.x, e0.y, e0.z, e0.w,  e1.x, e1.y, e1.z, e1.w,
                        e2.x, e2.y, e2.z, e2.w,  e3.x, e3.y, e3.z, e3.w};
        v16h bm;
        #pragma unroll
        for (int j = 0; j < 16; ++j) bm[j] = (_Float16)ev[j];

        // ---- ids for A matrix: element j <-> K = 16*half + j  (contiguous pixels)
        const int* ip = islab + pbase;
        v4i d0 = *(const v4i*)(ip + K0 + 0);
        v4i d1 = *(const v4i*)(ip + K0 + 4);
        v4i d2 = *(const v4i*)(ip + K0 + 8);
        v4i d3 = *(const v4i*)(ip + K0 + 12);
        int idv[16] = {d0.x, d0.y, d0.z, d0.w,  d1.x, d1.y, d1.z, d1.w,
                       d2.x, d2.y, d2.z, d2.w,  d3.x, d3.y, d3.z, d3.w};

        // ---- counts: one coalesced id per lane (issue before the burst so
        //      the ds_add fills slots while WMMAs drain)
        const int myid = islab[pbase + lane];
        atomicAdd(&lds_counts[myid], 1);

        // ---- prefetch next iteration's tiles (+256 pixels) into cache
        if (it + 1 < niters) {
            __builtin_prefetch(ep + WAVES * 32, 0, 1);   // global_prefetch_b8
            __builtin_prefetch(ip + WAVES * 32, 0, 1);
        }

        // ---- build all four A tiles
        v16h a0t, a1t, a2t, a3t;
        #pragma unroll
        for (int j = 0; j < 16; ++j) {
            const int idj = idv[j];
            a0t[j] = (idj == t0) ? one : zero;
            a1t[j] = (idj == t1) ? one : zero;
            a2t[j] = (idj == t2) ? one : zero;
            a3t[j] = (idj == t3) ? one : zero;
        }

        // Pin the four A tiles + B into distinct live VGPR tuples, and stop
        // the scheduler from sinking the one-hot builds into the WMMA burst.
        asm volatile("" : "+v"(a0t), "+v"(a1t), "+v"(a2t), "+v"(a3t), "+v"(bm));
#if __has_builtin(__builtin_amdgcn_sched_barrier)
        __builtin_amdgcn_sched_barrier(0);
#endif

        // ---- WMMA burst: four independent matrix ops, disjoint registers
        acc0 = __builtin_amdgcn_wmma_f32_16x16x32_f16(false, a0t, false, bm,
                                                      (short)0, acc0, false, false);
        acc1 = __builtin_amdgcn_wmma_f32_16x16x32_f16(false, a1t, false, bm,
                                                      (short)0, acc1, false, false);
        acc2 = __builtin_amdgcn_wmma_f32_16x16x32_f16(false, a2t, false, bm,
                                                      (short)0, acc2, false, false);
        acc3 = __builtin_amdgcn_wmma_f32_16x16x32_f16(false, a3t, false, bm,
                                                      (short)0, acc3, false, false);
    }

    // ---- scatter wave accumulators into LDS.
    // D layout: VGPR r, lane l -> (M = r + 8*half, N = col)
    #pragma unroll
    for (int r = 0; r < 8; ++r) {
        const int row = r + 8 * half;
        atomicAdd(&lds_sums[( 0 + row) * C_ + col], acc0[r]);
        atomicAdd(&lds_sums[(16 + row) * C_ + col], acc1[r]);
        atomicAdd(&lds_sums[(32 + row) * C_ + col], acc2[r]);
        atomicAdd(&lds_sums[(48 + row) * C_ + col], acc3[r]);
    }
    __syncthreads();

    // ---- flush per-block partials to global (few atomics per block)
    for (int i = tid; i < K_ * C_; i += THREADS)
        atomicAdd(&g_sums[slab * K_ * C_ + i], lds_sums[i]);
    if (tid < K_)
        atomicAdd(&g_counts[slab * K_ + tid], lds_counts[tid]);
}

// ---------------------------------------------------------------------------
// Kernel 2: finalize — 4*7*64 = 1792 frame-pair candidates, mean distance.
// ---------------------------------------------------------------------------
__global__ __launch_bounds__(256) void finalize_kernel(const float* __restrict__ sums,
                                                       const int*   __restrict__ counts,
                                                       float* __restrict__ out) {
    __shared__ float red_t[THREADS];
    __shared__ float red_n[THREADS];
    const int tid = threadIdx.x;
    float total = 0.0f, nvalid = 0.0f;

    const int NPAIR = B_ * (F_ - 1) * K_;
    for (int p = tid; p < NPAIR; p += THREADS) {
        const int k   = p & (K_ - 1);
        const int rem = p >> 6;
        const int fi  = rem % (F_ - 1);
        const int b   = rem / (F_ - 1);
        if (k == 0) continue;                 // arange(K) > 0
        const int s1 = (b * F_ + fi) * K_ + k;
        const int s2 = s1 + K_;               // next frame, same id
        const float c1 = (float)counts[s1];
        const float c2 = (float)counts[s2];
        if (c1 > 0.0f && c2 > 0.0f) {
            const float inv1 = 1.0f / fmaxf(c1, 1.0f);
            const float inv2 = 1.0f / fmaxf(c2, 1.0f);
            float d = 0.0f;
            #pragma unroll
            for (int c = 0; c < C_; ++c) {
                const float m1 = sums[s1 * C_ + c] * inv1;
                const float m2 = sums[s2 * C_ + c] * inv2;
                const float df = m1 - m2;
                d += df * df;
            }
            total  += d;
            nvalid += 1.0f;
        }
    }
    red_t[tid] = total;
    red_n[tid] = nvalid;
    __syncthreads();
    for (int s = THREADS / 2; s > 0; s >>= 1) {
        if (tid < s) { red_t[tid] += red_t[tid + s]; red_n[tid] += red_n[tid + s]; }
        __syncthreads();
    }
    if (tid == 0) out[0] = (red_n[0] > 0.0f) ? (red_t[0] / red_n[0]) : 0.0f;
}

// ---------------------------------------------------------------------------
extern "C" void kernel_launch(void* const* d_in, const int* in_sizes, int n_in,
                              void* d_out, int out_size, void* d_ws, size_t ws_size,
                              hipStream_t stream) {
    (void)in_sizes; (void)n_in; (void)out_size; (void)ws_size;

    const float* emb = (const float*)d_in[0];   // (4,8,16,256,256) f32
    const int*   ids = (const int*)d_in[1];     // (4,8,1,256,256) i32
    float*       out = (float*)d_out;           // scalar f32

    float* sums   = (float*)d_ws;                              // 32768 f32
    int*   counts = (int*)((char*)d_ws + WS_SUM_ELEMS * 4);    // 2048 i32

    const int zblocks = (WS_TOTAL_WORDS + 255) / 256;
    zero_ws_kernel<<<zblocks, 256, 0, stream>>>((unsigned*)d_ws, WS_TOTAL_WORDS);

    accum_kernel<<<SLABS * BLOCKS_PER_SLAB, THREADS, 0, stream>>>(emb, ids, sums, counts);

    finalize_kernel<<<1, THREADS, 0, stream>>>(sums, counts, out);
}
